// LSAAttention_62474594287824
// MI455X (gfx1250) — matrix operand order repaired and checked by reading, not compiled, not run on hardware
//
#include <hip/hip_runtime.h>
#include <hip/hip_bf16.h>

typedef __attribute__((ext_vector_type(16))) _Float16 v16h;
typedef __attribute__((ext_vector_type(8)))  _Float16 v8h;
typedef __attribute__((ext_vector_type(4)))  _Float16 v4h;
typedef __attribute__((ext_vector_type(8)))  float    v8f;
typedef __attribute__((ext_vector_type(4)))  float    v4f;

namespace {
constexpr int NH    = 6;
constexpr int HD    = 32;
constexpr int CD    = 192;          // channels
constexpr int OD    = 576;          // 3*C
constexpr int WGRID = 96;           // windows per row (wg)
constexpr int TG    = 128 * 96;     // windows per batch
constexpr int NTOK  = TG * 9;       // 110592 tokens per batch
constexpr int WPB   = 16;           // windows per block
constexpr int TPB   = WPB * 9;      // 144 tokens per block
constexpr int MT    = TPB / 16;     // 9 M-tiles
constexpr int NWAVE = 16;           // 512 threads -> 4 waves/SIMD32
constexpr int XS_S  = 200;          // f16 row stride (pad vs 192)
constexpr int Q_S   = 584;          // f16 row stride for QKV (pad vs 576)
constexpr int XS_OFF  = 0;
constexpr int QKV_OFF = TPB * XS_S * 2;              // 57600 B
constexpr int SCR_OFF = QKV_OFF + TPB * Q_S * 2;     // 225792 B
constexpr int LDS_BYTES = SCR_OFF + NWAVE * 2048;    // 258560 B (<320KB/WGP)
}

__device__ __forceinline__ v8f wmma16(v16h a, v16h b, v8f c) {
  // v_wmma_f32_16x16x32_f16 : D = A(16x32 f16) * B(32x16 f16) + C(f32)
  return __builtin_amdgcn_wmma_f32_16x16x32_f16(false, a, false, b, (short)0, c, false, false);
}

// A-operand gather (16-bit A 16x32): lane m=l&15, hh=l>>4
// elems 0..15 <-> K = 16*hh + 0..15 contiguous (at p[0..15])
__device__ __forceinline__ v16h lds_loadA(const _Float16* p) {
  v16h r;
  ((v8h*)&r)[0] = *(const v8h*)(p);
  ((v8h*)&r)[1] = *(const v8h*)(p + 8);
  return r;
}

// B-operand gather: lane n=l&15 holds K = 16*hh + 0..15 contiguous
__device__ __forceinline__ v16h lds_loadB(const _Float16* p) {
  v16h r;
  ((v8h*)&r)[0] = *(const v8h*)(p);
  ((v8h*)&r)[1] = *(const v8h*)(p + 8);
  return r;
}

// B-operand from f32 weights in global (row-major [O][C]); p pre-offset to
// row (n0 + lane&15), col c0 + 16*hh : 16 contiguous floats -> 16 f16 elems.
__device__ __forceinline__ v16h glb_loadB_f32(const float* p) {
  v4f a = *(const v4f*)(p);
  v4f b = *(const v4f*)(p + 4);
  v4f c = *(const v4f*)(p + 8);
  v4f d = *(const v4f*)(p + 12);
  v16h r;
#pragma unroll
  for (int i = 0; i < 4; ++i) {
    r[i]      = (_Float16)a[i];
    r[4 + i]  = (_Float16)b[i];
    r[8 + i]  = (_Float16)c[i];
    r[12 + i] = (_Float16)d[i];
  }
  return r;
}

__global__ __launch_bounds__(512)
void lsa_fused(const float* __restrict__ x,
               const float* __restrict__ Wqkv,
               const float* __restrict__ Wproj,
               const float* __restrict__ bproj,
               float* __restrict__ out) {
  extern __shared__ char smem[];
  _Float16* Xs   = (_Float16*)(smem + XS_OFF);
  _Float16* QKVs = (_Float16*)(smem + QKV_OFF);
  _Float16* Att  = (_Float16*)(smem + XS_OFF);   // aliases Xs (X dead after GEMM1)

  const int tid  = threadIdx.x;
  const int lane = tid & 31;
  const int wv   = tid >> 5;          // wave id (16 waves)
  const int n15  = lane & 15;
  const int hh   = lane >> 4;

  const int w0    = blockIdx.x * WPB;
  const int batch = w0 / TG;
  const int w     = w0 - batch * TG;
  const int arow  = w / WGRID;        // hg index
  const int b2    = w - arow * WGRID; // wg base (multiple of 16)

  const float* xb = x + (size_t)batch * NTOK * CD;

  // ---- stage X tile: 3 chunks of 48 contiguous tokens x 192ch, f32 -> f16 ----
#pragma unroll 1
  for (int f = tid; f < TPB * CD / 4; f += 512) {
    int chunk = f / (48 * CD / 4);
    int v4i   = f - chunk * (48 * CD / 4);
    int flat  = v4i * 4;
    size_t gbase = (size_t)(((arow * 3 + chunk) * WGRID + b2) * 3) * CD;
    v4f dv = *(const v4f*)(xb + gbase + flat);
    int tt = chunk * 48 + flat / CD;
    int c  = flat - (flat / CD) * CD;
    v4h h4;
#pragma unroll
    for (int i = 0; i < 4; ++i) h4[i] = (_Float16)dv[i];
    *(v4h*)(Xs + tt * XS_S + c) = h4;
  }
  __syncthreads();

  // ---- GEMM1: QKV(144x576) = X(144x192) @ Wqkv^T, f16 WMMA, f32 acc ----
  for (int nt = wv; nt < OD / 16; nt += NWAVE) {
    v16h Bw[6];
#pragma unroll
    for (int kc = 0; kc < 6; ++kc)
      Bw[kc] = glb_loadB_f32(Wqkv + (size_t)(nt * 16 + n15) * CD + kc * 32 + hh * 16);
    for (int mt = 0; mt < MT; ++mt) {
      const _Float16* ap = Xs + (mt * 16 + n15) * XS_S + hh * 16;
      v16h Aop[6];                               // all 12 ds_load_b128 in flight
#pragma unroll
      for (int kc = 0; kc < 6; ++kc) Aop[kc] = lds_loadA(ap + kc * 32);
      v8f acc0, acc1;
#pragma unroll
      for (int r = 0; r < 8; ++r) { acc0[r] = 0.0f; acc1[r] = 0.0f; }
#pragma unroll
      for (int kc = 0; kc < 6; kc += 2) {        // two independent WMMA chains
        acc0 = wmma16(Aop[kc],     Bw[kc],     acc0);
        acc1 = wmma16(Aop[kc + 1], Bw[kc + 1], acc1);
      }
      v8f acc = acc0 + acc1;
      int o = nt * 16 + n15;
#pragma unroll
      for (int r = 0; r < 8; ++r)                 // D: M=r+8*hh, N=n15
        QKVs[(mt * 16 + r + 8 * hh) * Q_S + o] = (_Float16)acc[r];
    }
  }
  __syncthreads();

  // ---- attention: one (window, head) per wave-task; 3 WMMAs + softmax ----
  {
    float*    Ssc = (float*)(smem + SCR_OFF + wv * 2048);           // 16x16 f32
    _Float16* Pb  = (_Float16*)(smem + SCR_OFF + wv * 2048 + 1024); // 16x32 f16
    const float scale = 0.17677669529663687f;    // 1/sqrt(32)
    for (int task = wv; task < WPB * NH; task += NWAVE) {
      int wl = task / NH;
      int h  = task - wl * NH;
      auto trow = [&](int q) {                   // window token -> tile row
        if (q > 8) q = 8;                        // clamp pad rows (values unused)
        int qi = q / 3;
        return qi * 48 + wl * 3 + (q - qi * 3);
      };
      const int myrow = trow(n15);
      // S = Q @ K^T  (A rows = queries, B cols = keys, K-dim = 32 head dims)
      v16h Aq = lds_loadA(QKVs + myrow * Q_S + h * HD + hh * 16);
      v16h Bk = lds_loadB(QKVs + myrow * Q_S + CD + h * HD + hh * 16);
      v8f S;
#pragma unroll
      for (int r = 0; r < 8; ++r) S[r] = 0.0f;
      S = wmma16(Aq, Bk, S);
#pragma unroll
      for (int r = 0; r < 8; ++r) Ssc[(r + 8 * hh) * 16 + n15] = S[r] * scale;
      // zero P (kills pad rows/cols so no NaN reaches P@V)
      {
        v8h z;
#pragma unroll
        for (int i = 0; i < 8; ++i) z[i] = (_Float16)0.0f;
        ((v8h*)Pb)[lane * 2]     = z;
        ((v8h*)Pb)[lane * 2 + 1] = z;
      }
      if (lane < 9) {                            // softmax over 9 valid cols
        float rv[9];
        float mx = -3.0e38f;
#pragma unroll
        for (int k = 0; k < 9; ++k) { rv[k] = Ssc[lane * 16 + k]; mx = fmaxf(mx, rv[k]); }
        float sum = 0.0f;
#pragma unroll
        for (int k = 0; k < 9; ++k) { rv[k] = __expf(rv[k] - mx); sum += rv[k]; }
        float inv = 1.0f / sum;
#pragma unroll
        for (int k = 0; k < 9; ++k) Pb[lane * 32 + k] = (_Float16)(rv[k] * inv);
      }
      // O = P @ V   (K-dim padded to 32 with zeros in P)
      v16h Ap = lds_loadA(Pb + n15 * 32 + hh * 16);
#pragma unroll
      for (int nt2 = 0; nt2 < 2; ++nt2) {
        int dcol = n15 + nt2 * 16;
        v16h Bv;
#pragma unroll
        for (int i = 0; i < 16; ++i) Bv[i] = (_Float16)0.0f;
        if (hh == 0) {                           // lane elems e <-> K = e (keys 0..8)
#pragma unroll
          for (int k = 0; k < 9; ++k)
            Bv[k] = QKVs[trow(k) * Q_S + 2 * CD + h * HD + dcol];
        }                                        // hh==1 lanes: K=16..31 -> all zero
        v8f O;
#pragma unroll
        for (int r = 0; r < 8; ++r) O[r] = 0.0f;
        O = wmma16(Ap, Bv, O);
#pragma unroll
        for (int r = 0; r < 8; ++r) {
          int m2 = r + 8 * hh;
          if (m2 < 9) Att[trow(m2) * XS_S + h * HD + dcol] = (_Float16)O[r];
        }
      }
    }
  }
  __syncthreads();

  // ---- GEMM2: out = Att(144x192) @ Wproj^T + bproj, fp32 to d_out ----
  // 12 n-tiles x 4 m-groups = 48 chunks -> exactly 3 per wave
  for (int ch = wv; ch < 48; ch += NWAVE) {
    int nt = ch % 12;
    int mg = ch / 12;
    v16h Bw[6];
#pragma unroll
    for (int kc = 0; kc < 6; ++kc)
      Bw[kc] = glb_loadB_f32(Wproj + (size_t)(nt * 16 + n15) * CD + kc * 32 + hh * 16);
    int o = nt * 16 + n15;
    float bias = bproj[o];
    const int mlo = mg * 2;                      // groups: [0,2)[2,4)[4,6)[6,9)
    const int mhi = (mg == 3) ? MT : (mg * 2 + 2);
    for (int mt = mlo; mt < mhi; ++mt) {
      const _Float16* ap = Att + (mt * 16 + n15) * XS_S + hh * 16;
      v16h Aop[6];
#pragma unroll
      for (int kc = 0; kc < 6; ++kc) Aop[kc] = lds_loadA(ap + kc * 32);
      v8f acc0, acc1;
#pragma unroll
      for (int r = 0; r < 8; ++r) { acc0[r] = 0.0f; acc1[r] = 0.0f; }
#pragma unroll
      for (int kc = 0; kc < 6; kc += 2) {
        acc0 = wmma16(Aop[kc],     Bw[kc],     acc0);
        acc1 = wmma16(Aop[kc + 1], Bw[kc + 1], acc1);
      }
      v8f acc = acc0 + acc1;
#pragma unroll
      for (int r = 0; r < 8; ++r) {
        int tt = mt * 16 + r + 8 * hh;
        int ic = tt / 48;
        int u  = tt - ic * 48;
        size_t ng = (size_t)(((arow * 3 + ic) * WGRID + b2) * 3) + u;
        out[((size_t)batch * NTOK + ng) * CD + o] = acc[r] + bias;
      }
    }
  }
}

extern "C" void kernel_launch(void* const* d_in, const int* in_sizes, int n_in,
                              void* d_out, int out_size, void* d_ws, size_t ws_size,
                              hipStream_t stream) {
  (void)in_sizes; (void)n_in; (void)out_size; (void)d_ws; (void)ws_size;
  const float* x     = (const float*)d_in[0];
  const float* Wqkv  = (const float*)d_in[1];
  const float* Wproj = (const float*)d_in[2];
  const float* bproj = (const float*)d_in[3];
  float* out = (float*)d_out;

  hipFuncSetAttribute(reinterpret_cast<const void*>(lsa_fused),
                      hipFuncAttributeMaxDynamicSharedMemorySize, LDS_BYTES);

  dim3 grid(2 * TG / WPB);   // 24576 windows / 16 = 1536 workgroups
  lsa_fused<<<grid, 512, LDS_BYTES, stream>>>(x, Wqkv, Wproj, bproj, out);
}
